// Tno_26620207301198
// MI455X (gfx1250) — matrix operand order repaired and checked by reading, not compiled, not run on hardware
//
#include <hip/hip_runtime.h>
#include <hip/hip_bf16.h>

// ---------------------------------------------------------------------------
// TNO global convolution for MI455X (gfx1250), compile-only target.
//
//   out[b,t,d] = sum_{s=0..4095} x[b,s,d] * a[(t-s) mod 8192, d]
//
// K4 computes, per channel d and 1024-wide t-chunk, the Toeplitz GEMM with
// v_wmma_f32_16x16x32_bf16:
//   A operand (16x32)  = x tile, rows = (batch, t-offset-group j), K = s
//                        -> two aligned ds_load_b128 per tile
//   B operand (32x16)  = Toeplitz tile from a reversed *linear* window in LDS
//                        (two parity-shifted copies) -> 8 aligned ds_load_b32
//   C tile (16x16 f32) = out[b, t0 + n + 16j] partials
// B tiles slide along the K loop (B(q,i)==B(q-1,i-1)); a 4-step unrolled
// group with permuted register names gives 1 fresh B + 1 fresh A + 4 WMMA per
// K step and no register rotation copies.
// ---------------------------------------------------------------------------

typedef __attribute__((ext_vector_type(16))) __bf16 v16bf;
typedef __attribute__((ext_vector_type(8)))  float  v8f;

#define B_SZ      8
#define N_SEQ     4096
#define D_CH      512
#define L_SEQ     8192
#define T_CHUNK   1024

// LDS layout (bytes) for K4
#define WIN_N     5136                 // reversed a-window elements
#define WRA_OFF   0                    // copy A: wr[i] = a[(hi - i) & 8191]
#define WRB_OFF   10304                // copy B: wr[i] = a[(hi-1-i) & 8191]
#define XL_OFF    20608                // padded x rows
#define XROW_EL   4160                 // 16 front pad + 4096 + 32 back pad + 16
#define XROW_BY   8320                 // bytes per row (16B multiple)
#define SMEM_BY   (XL_OFF + B_SZ * XROW_BY)   // 87168

// ------------------------------- K1: RPE MLP -------------------------------

__device__ inline float wave_sum(float v) {
#pragma unroll
  for (int m = 16; m > 0; m >>= 1) v += __shfl_xor(v, m, 32);
  return v;
}

__global__ void __launch_bounds__(256)
mlp_kernel(const float* __restrict__ rpe, const float* __restrict__ pos_w,
           const float* __restrict__ pos_b, const float* __restrict__ ln_g,
           const float* __restrict__ ln_b, const float* __restrict__ lw,
           const float* __restrict__ lb, const float* __restrict__ og,
           const float* __restrict__ ob, const float* __restrict__ ow,
           const float* __restrict__ obias, const float* __restrict__ coef,
           float* __restrict__ a_out) {
  const int lane = threadIdx.x & 31;
  const int wave = threadIdx.x >> 5;
  const int l = blockIdx.x * 8 + wave;

  float h = rpe[l] * pos_w[lane] + pos_b[lane];

#pragma unroll
  for (int i = 0; i < 3; ++i) {
    float mu  = wave_sum(h) * (1.0f / 32.0f);
    float xm  = h - mu;
    float var = wave_sum(xm * xm) * (1.0f / 32.0f);
    float t   = xm * rsqrtf(var + 1e-5f) * ln_g[i * 32 + lane] + ln_b[i * 32 + lane];
    float r   = fmaxf(t, 0.0f);
    float acc = lb[i * 32 + lane];
#pragma unroll
    for (int k = 0; k < 32; ++k) {
      float rk = __shfl(r, k, 32);
      acc = fmaf(rk, lw[(i * 32 + k) * 32 + lane], acc);
    }
    h = h + acc;
  }

  float mu  = wave_sum(h) * (1.0f / 32.0f);
  float xm  = h - mu;
  float var = wave_sum(xm * xm) * (1.0f / 32.0f);
  float t   = xm * rsqrtf(var + 1e-5f) * og[lane] + ob[lane];
  float r   = fmaxf(t, 0.0f);

  float bs = (lane < 16) ? obias[lane] : 0.0f;
#pragma unroll
  for (int k = 0; k < 32; ++k) {
    float rk = __shfl(r, k, 32);
    if (lane < 16) bs = fmaf(rk, ow[k * 16 + lane], bs);
  }

  float base[16];
#pragma unroll
  for (int rr = 0; rr < 16; ++rr) base[rr] = __shfl(bs, rr, 32);

  for (int d = lane; d < D_CH; d += 32) {
    float acc = 0.0f;
#pragma unroll
    for (int rr = 0; rr < 16; ++rr) acc = fmaf(base[rr], coef[rr * D_CH + d], acc);
    a_out[(size_t)l * D_CH + d] = acc;
  }
}

// ------------------------- K2/K3/K5: layout shuffles -----------------------

__global__ void __launch_bounds__(256)
a_transpose_kernel(const float* __restrict__ a, __bf16* __restrict__ aT) {
  __shared__ float tile[32][33];
  const int dbase = blockIdx.x * 32, lbase = blockIdx.y * 32;
  const int tx = threadIdx.x, ty = threadIdx.y;
#pragma unroll
  for (int i = 0; i < 4; ++i)
    tile[ty + 8 * i][tx] = a[(size_t)(lbase + ty + 8 * i) * D_CH + dbase + tx];
  __syncthreads();
#pragma unroll
  for (int i = 0; i < 4; ++i)
    aT[(size_t)(dbase + ty + 8 * i) * L_SEQ + lbase + tx] =
        (__bf16)tile[tx][ty + 8 * i];
}

__global__ void __launch_bounds__(256)
x_transpose_kernel(const float* __restrict__ x, __bf16* __restrict__ xT) {
  __shared__ float tile[32][33];
  const int b = blockIdx.z;
  const int dbase = blockIdx.x * 32, sbase = blockIdx.y * 32;
  const int tx = threadIdx.x, ty = threadIdx.y;
#pragma unroll
  for (int i = 0; i < 4; ++i)
    tile[ty + 8 * i][tx] =
        x[((size_t)b * N_SEQ + sbase + ty + 8 * i) * D_CH + dbase + tx];
  __syncthreads();
#pragma unroll
  for (int i = 0; i < 4; ++i)
    xT[((size_t)(dbase + ty + 8 * i) * B_SZ + b) * N_SEQ + sbase + tx] =
        (__bf16)tile[tx][ty + 8 * i];
}

__global__ void __launch_bounds__(256)
out_transpose_kernel(const float* __restrict__ outT, float* __restrict__ out) {
  __shared__ float tile[32][33];
  const int b = blockIdx.z;
  const int dbase = blockIdx.x * 32, tbase = blockIdx.y * 32;
  const int tx = threadIdx.x, ty = threadIdx.y;
#pragma unroll
  for (int i = 0; i < 4; ++i)
    tile[ty + 8 * i][tx] =
        outT[((size_t)(dbase + ty + 8 * i) * B_SZ + b) * N_SEQ + tbase + tx];
  __syncthreads();
#pragma unroll
  for (int i = 0; i < 4; ++i)
    out[((size_t)b * N_SEQ + tbase + ty + 8 * i) * D_CH + dbase + tx] =
        tile[tx][ty + 8 * i];
}

// --------------------------- K4: Toeplitz WMMA GEMM ------------------------

__device__ inline v8f wmma_bf16(v16bf a, v16bf b, v8f c) {
  return __builtin_amdgcn_wmma_f32_16x16x32_bf16(false, a, false, b, (short)0, c,
                                                 false, false);
}

// B operand (Toeplitz): 8 aligned ds_load_b32 at immediate offsets.
__device__ inline v16bf load_b_tile(const char* p) {
  union { v16bf v; unsigned u[8]; } B;
#pragma unroll
  for (int w = 0; w < 8; ++w) B.u[w] = *(const unsigned*)(p + 4 * w);
  return B.v;
}

// A operand (x data): two aligned ds_load_b128 (runs K=baseK.. and +8).
__device__ inline v16bf load_a_tile(const char* p) {
  union { v16bf v; uint4 q[2]; } A;
  A.q[0] = *(const uint4*)(p);
  A.q[1] = *(const uint4*)(p + 16);
  return A.v;
}

__global__ void __launch_bounds__(256)
tno_conv_kernel(const __bf16* __restrict__ aT, const __bf16* __restrict__ xT,
                float* __restrict__ outT) {
  extern __shared__ char smem[];
  __bf16* wrA = (__bf16*)(smem + WRA_OFF);
  __bf16* wrB = (__bf16*)(smem + WRB_OFF);
  float* stage = (float*)(smem + XL_OFF);   // reused after the K loop (32KB)

  const int d     = blockIdx.y;
  const int tbase = blockIdx.x * T_CHUNK;
  const int tid   = threadIdx.x;
  const int hi    = tbase + T_CHUNK - 1;

  __builtin_prefetch(outT + ((size_t)d * B_SZ) * N_SEQ + tbase, 0, 1);

  // ---- fill reversed linear a-window (two parity copies) ----
  {
    const __bf16* aTd = aT + (size_t)d * L_SEQ;
    for (int i = tid; i < WIN_N; i += 256) {
      wrA[i] = aTd[(hi - i) & (L_SEQ - 1)];
      wrB[i] = aTd[(hi - 1 - i) & (L_SEQ - 1)];
    }
  }
  // ---- fill padded x rows: [16 zeros | x_d[b][0..4095] | 32 zeros] ----
#pragma unroll
  for (int b = 0; b < B_SZ; ++b) {
    const uint4* src = (const uint4*)(xT + ((size_t)d * B_SZ + b) * N_SEQ);
    char* row = smem + XL_OFF + b * XROW_BY;
    for (int i = tid; i < 512; i += 256)
      *(uint4*)(row + 32 + 16 * i) = src[i];
  }
  if (tid < B_SZ) {
    char* row = smem + XL_OFF + tid * XROW_BY;
    uint4 z = {0u, 0u, 0u, 0u};
    *(uint4*)(row + 0)  = z;
    *(uint4*)(row + 16) = z;
#pragma unroll
    for (int k = 0; k < 6; ++k) *(uint4*)(row + 8224 + 16 * k) = z;
  }
  __syncthreads();

  const int lane = tid & 31;
  const int wave = tid >> 5;
  // A-operand lane mapping (rows m = (b, j))
  const int bA    = lane & 7;
  const int jA    = (lane >> 3) & 1;
  const int baseK = (lane < 16) ? 0 : 16;
  // B-operand lane mapping (columns n, K halves)
  const int nIdx = lane & 15;
  const int kb   = (lane < 16) ? 0 : 16;

  // A pointer at K step i: byte = rowbase + 64*i + 2*(16*jA + baseK)
  const char* ap0 = smem + XL_OFF + bA * XROW_BY + 2 * (16 * jA + baseK);
  const char* ap  = ap0 + 64;  // i = 1

  // B fresh-tile (q=0) index at step i: R(i) = hi - tw - 16 + 32*i + kb - n
  const int tw = tbase + wave * 128;
  const int R1 = (hi - tw) + 16 + kb - nIdx;           // i = 1
  const int par = R1 & 1;
  const char* bcopy = smem + (par ? WRB_OFF : WRA_OFF);
  const char* bp = bcopy + 2 * (R1 - par);             // i = 1, 4B aligned

  v8f c0 = {}, c1 = {}, c2 = {}, c3 = {};

  // prologue: B(q, i=0) at byte bp - 64 - 64*q
  v16bf S0 = load_b_tile(bp - 64);
  v16bf S1 = load_b_tile(bp - 128);
  v16bf S2 = load_b_tile(bp - 192);
  v16bf S3 = load_b_tile(bp - 256);

  // ---- K step i = 0 ----
  {
    v16bf Av = load_a_tile(ap0);
    c0 = wmma_bf16(Av, S0, c0);
    c1 = wmma_bf16(Av, S1, c1);
    c2 = wmma_bf16(Av, S2, c2);
    c3 = wmma_bf16(Av, S3, c3);
  }

  // ---- K steps i = 1..128 : 32 groups of 4 with register renaming ----
#define CONV_STEP(BF, BQ1, BQ2, BQ3, OFF)                                    \
  {                                                                          \
    BF = load_b_tile(bp + (OFF)*64);                                         \
    v16bf Av = load_a_tile(ap + (OFF)*64);                                   \
    c0 = wmma_bf16(Av, BF, c0);                                              \
    c1 = wmma_bf16(Av, BQ1, c1);                                             \
    c2 = wmma_bf16(Av, BQ2, c2);                                             \
    c3 = wmma_bf16(Av, BQ3, c3);                                             \
  }
#pragma unroll 1
  for (int g = 0; g < 32; ++g) {
    CONV_STEP(S3, S0, S1, S2, 0)
    CONV_STEP(S2, S3, S0, S1, 1)
    CONV_STEP(S1, S2, S3, S0, 2)
    CONV_STEP(S0, S1, S2, S3, 3)
    bp += 256;
    ap += 256;
  }
#undef CONV_STEP

  __syncthreads();  // done reading x/window; reuse LDS as stage

  // C layout: VGPR r -> b = r; lanes>=16 -> j = 1; n = lane & 15
  const int jj = (lane >> 4) & 1;
#define STAGE_TILE(Cq, q)                                                    \
  {                                                                          \
    _Pragma("unroll") for (int r = 0; r < 8; ++r) {                          \
      stage[r * T_CHUNK + wave * 128 + (q)*32 + nIdx + 16 * jj] = (Cq)[r];   \
    }                                                                        \
  }
  STAGE_TILE(c0, 0)
  STAGE_TILE(c1, 1)
  STAGE_TILE(c2, 2)
  STAGE_TILE(c3, 3)
#undef STAGE_TILE
  __syncthreads();

  // coalesced write: outT[d][b][tbase .. tbase+1024)
  for (int f = tid; f < 2048; f += 256) {
    int f4 = f * 4;
    int b = f4 >> 10;
    int tloc = f4 & 1023;
    *(uint4*)(outT + ((size_t)d * B_SZ + b) * N_SEQ + tbase + tloc) =
        *(const uint4*)(stage + f4);
  }
}

// ------------------------------- launcher ----------------------------------

extern "C" void kernel_launch(void* const* d_in, const int* in_sizes, int n_in,
                              void* d_out, int out_size, void* d_ws, size_t ws_size,
                              hipStream_t stream) {
  const float* x     = (const float*)d_in[0];
  const float* rpe   = (const float*)d_in[1];
  const float* pos_w = (const float*)d_in[2];
  const float* pos_b = (const float*)d_in[3];
  const float* ln_g  = (const float*)d_in[4];
  const float* ln_b  = (const float*)d_in[5];
  const float* lw    = (const float*)d_in[6];
  const float* lb    = (const float*)d_in[7];
  const float* og    = (const float*)d_in[8];
  const float* ob    = (const float*)d_in[9];
  const float* ow    = (const float*)d_in[10];
  const float* obias = (const float*)d_in[11];
  const float* coef  = (const float*)d_in[12];

  char* ws = (char*)d_ws;
  float*  a_f32 = (float*)ws;                        // 16 MB
  __bf16* aT    = (__bf16*)(ws + (16ull << 20));     //  8 MB
  __bf16* xT    = (__bf16*)(ws + (24ull << 20));     // 32 MB
  float*  outT  = (float*)(ws + (56ull << 20));      // 64 MB (total 120 MB)

  mlp_kernel<<<L_SEQ / 8, 256, 0, stream>>>(rpe, pos_w, pos_b, ln_g, ln_b, lw,
                                            lb, og, ob, ow, obias, coef, a_f32);
  dim3 bT(32, 8);
  a_transpose_kernel<<<dim3(D_CH / 32, L_SEQ / 32), bT, 0, stream>>>(a_f32, aT);
  x_transpose_kernel<<<dim3(D_CH / 32, N_SEQ / 32, B_SZ), bT, 0, stream>>>(x, xT);

  (void)hipFuncSetAttribute((const void*)tno_conv_kernel,
                            hipFuncAttributeMaxDynamicSharedMemorySize, SMEM_BY);
  tno_conv_kernel<<<dim3(N_SEQ / T_CHUNK, D_CH), 256, SMEM_BY, stream>>>(aT, xT,
                                                                         outT);

  out_transpose_kernel<<<dim3(D_CH / 32, N_SEQ / 32, B_SZ), bT, 0, stream>>>(
      outT, (float*)d_out);
}
